// MultiHeadAttention_8881992368623
// MI455X (gfx1250) — matrix operand order repaired and checked by reading, not compiled, not run on hardware
//
#include <hip/hip_runtime.h>

typedef __attribute__((ext_vector_type(16))) __bf16         v16bf;
typedef __attribute__((ext_vector_type(16))) unsigned short us16;
typedef __attribute__((ext_vector_type(4)))  unsigned int   u32x4;
typedef __attribute__((ext_vector_type(8)))  float          v8f;
typedef __attribute__((__vector_size__(16))) int            v4i;

#define NB    4
#define SEQ   1024
#define EV    768      // 8*8*12
#define NH    12
#define HD    64
#define ATT_SCALE 0.125f   // folded into Q at projection time (exact: 2^-3)
#define NQKV  3145728      // NB*NH*SEQ*HD

#if __has_builtin(__builtin_amdgcn_global_load_async_to_lds_b128)
#define HAVE_ASYNC_LDS 1
typedef __attribute__((address_space(1))) v4i as1_v4i;
typedef __attribute__((address_space(3))) v4i as3_v4i;
#endif

union UF16 { us16 v; u32x4 q[2]; };
union UV16 { u32x4 q; unsigned short h[8]; };

__device__ __forceinline__ unsigned short f2bf(float f) {
    unsigned int u = __float_as_uint(f);
    u += 0x7fffu + ((u >> 16) & 1u);          // round-to-nearest-even
    return (unsigned short)(u >> 16);
}

__device__ __forceinline__ v8f wmma_bf16(us16 a, us16 b, v8f c) {
    return __builtin_amdgcn_wmma_f32_16x16x32_bf16(
        false, __builtin_bit_cast(v16bf, a),
        false, __builtin_bit_cast(v16bf, b),
        (short)0, c, false, false);
}

// ---------------------------------------------------------------------------
// Kernel 1: Tucker projections for Q,K,V + split-heads, fp32 -> bf16
// out layout: [b][head(12)][seq(1024)][d(64)], d = x*16 + y*4 + z
// Q is pre-scaled by ATT_SCALE (exact power of two in bf16).
// ---------------------------------------------------------------------------
__global__ void __launch_bounds__(256) proj_qkv_kernel(
    const float* __restrict__ x,
    const float* __restrict__ w1q, const float* __restrict__ w2q, const float* __restrict__ w3q,
    const float* __restrict__ w1k, const float* __restrict__ w2k, const float* __restrict__ w3k,
    const float* __restrict__ w1v, const float* __restrict__ w2v, const float* __restrict__ w3v,
    unsigned short* __restrict__ qh, unsigned short* __restrict__ kh,
    unsigned short* __restrict__ vh)
{
    __shared__ float xv[EV], t1[EV], t2[EV];
    __shared__ float ws1[64], ws2[64], ws3[144];

    const int vox = blockIdx.x;               // b*1024 + p*32 + q
    const int b   = vox >> 10;
    const int seq = vox & 1023;
    const int tid = threadIdx.x;

    for (int e = tid; e < EV; e += 256) xv[e] = x[(size_t)vox * EV + e];

    const float* W1[3] = {w1q, w1k, w1v};
    const float* W2[3] = {w2q, w2k, w2v};
    const float* W3[3] = {w3q, w3k, w3v};
    unsigned short* DST[3] = {qh, kh, vh};

    for (int pi = 0; pi < 3; ++pi) {
        __syncthreads();
        if (tid < 64)  { ws1[tid] = W1[pi][tid]; ws2[tid] = W2[pi][tid]; }
        if (tid < 144) { ws3[tid] = W3[pi][tid]; }
        __syncthreads();

        // stage 1: t1[x,w,c] = sum_h xv[h,w,c] * w1[h,x]
        for (int e = tid; e < EV; e += 256) {
            int xx = e / 96, rem = e - xx * 96;
            float acc = 0.f;
            #pragma unroll
            for (int h = 0; h < 8; ++h) acc += xv[h * 96 + rem] * ws1[h * 8 + xx];
            t1[e] = acc;
        }
        __syncthreads();

        // stage 2: t2[x,y,c] = sum_w t1[x,w,c] * w2[w,y]
        for (int e = tid; e < EV; e += 256) {
            int xx = e / 96, r = e - xx * 96;
            int yy = r / 12, c = r - yy * 12;
            float acc = 0.f;
            #pragma unroll
            for (int w = 0; w < 8; ++w) acc += t1[xx * 96 + w * 12 + c] * ws2[w * 8 + yy];
            t2[e] = acc;
        }
        __syncthreads();

        // stage 3: out[x,y,z] = sum_c t2[x,y,c] * w3[c,z]; fused split-heads store
        unsigned short* dst = DST[pi];
        const float post = (pi == 0) ? ATT_SCALE : 1.0f;
        for (int e = tid; e < EV; e += 256) {
            int e1 = e / 96, r = e - e1 * 96;
            int e2 = r / 12, e3 = r - e2 * 12;
            float acc = 0.f;
            #pragma unroll
            for (int c = 0; c < 12; ++c) acc += t2[e1 * 96 + e2 * 12 + c] * ws3[c * 12 + e3];
            int hx = e1 >> 1, h1 = e1 & 1;
            int hy = e2 >> 1, h2 = e2 & 1;
            int hz = e3 / 3,  h3 = e3 - hz * 3;
            int head = (h1 * 2 + h2) * 3 + h3;
            int d = hx * 16 + hy * 4 + hz;
            dst[(((size_t)(b * NH + head) * SEQ) + seq) * HD + d] = f2bf(acc * post);
        }
    }
}

// ---------------------------------------------------------------------------
// Kernel 2: flash attention. One (b,head) per blockIdx.x, 64 q-rows per block,
// 4 waves x 16 q-rows. 64-key chunks: 8 QK^T + 2 rowsum + 8 PV WMMAs/iter.
// K staged via GLOBAL_LOAD_ASYNC_TO_LDS (ASYNCcnt). Within each 64-key chunk
// the key order is permuted (kk = 4*c + t) identically for P columns and V
// rows, so P and V staging use packed b64 LDS stores.
// ---------------------------------------------------------------------------
__global__ void __launch_bounds__(128) attn_kernel(
    const unsigned short* __restrict__ qh,
    const unsigned short* __restrict__ kh,
    const unsigned short* __restrict__ vh,
    float* __restrict__ oh)
{
    __shared__ __align__(16) unsigned short Ks[64 * 64];       // [k][d]
    __shared__ __align__(16) unsigned short Vts[64 * 72];      // [d][kk], padded stride 72
    __shared__ __align__(16) unsigned short Ps[4 * 16 * 72];   // per-wave [row][kk], stride 72

    const int bh   = blockIdx.x;          // b*12 + head
    const int qblk = blockIdx.y;          // 0..15
    const int tid  = threadIdx.x;
    const int wv   = tid >> 5;
    const int lane = tid & 31;
    const int half = lane >> 4;
    const int l15  = lane & 15;

    const unsigned short* qbase = qh + (size_t)bh * SEQ * HD;
    const unsigned short* kbase = kh + (size_t)bh * SEQ * HD;
    const unsigned short* vbase = vh + (size_t)bh * SEQ * HD;

    // Q A-fragments (ISA 16-bit A layout): lane<16 row=lane, K{0..7,16..23};
    // lane>=16 row=lane-16, K{8..15,24..31}. Two K=32 chunks cover d=64.
    const int qrow = qblk * 64 + wv * 16 + l15;
    us16 qa[2];
    {
        const unsigned short* qp = qbase + (size_t)qrow * HD;
        #pragma unroll
        for (int c = 0; c < 2; ++c) {
            int k0 = c * 32 + half * 8;
            UF16 t;
            t.q[0] = *(const u32x4*)(qp + k0);
            t.q[1] = *(const u32x4*)(qp + k0 + 16);
            qa[c] = t.v;
        }
    }

    // ones B-fragment (bf16 1.0) for WMMA row-sum
    us16 ones;
    #pragma unroll
    for (int i = 0; i < 16; ++i) ones[i] = 0x3F80;

    float m[8], lsum[8];
    v8f o[4];
    #pragma unroll
    for (int i = 0; i < 8; ++i) { m[i] = -1e30f; lsum[i] = 0.f; }
    #pragma unroll
    for (int ng = 0; ng < 4; ++ng) { v8f z = {}; o[ng] = z; }

    // V-staging mapping (fixed per thread): column-in-tile c, feature group d0
    const int vc  = tid & 15;
    const int vd0 = (tid >> 4) * 8;

    for (int ch = 0; ch < SEQ / 64; ++ch) {
        __syncthreads();

        // ---- stage K chunk: async global->LDS (8KB) ----------------------
        const unsigned short* ksrc = kbase + (size_t)ch * 64 * HD;
#ifdef HAVE_ASYNC_LDS
        for (int i = tid; i < 512; i += 128) {
            __builtin_amdgcn_global_load_async_to_lds_b128(
                (as1_v4i*)(ksrc + i * 8), (as3_v4i*)(Ks + i * 8), 0, 0);
        }
#else
        {
            const u32x4* src = (const u32x4*)ksrc;
            u32x4* dst = (u32x4*)Ks;
            for (int i = tid; i < 512; i += 128) dst[i] = src[i];
        }
#endif
        // ---- stage V chunk transposed+permuted: Vts[d][4c+t] = V[16t+c][d]
        {
            const unsigned short* vsrc = vbase + (size_t)ch * 64 * HD;
            UV16 r0, r1, r2, r3;
            r0.q = *(const u32x4*)(vsrc + (0 * 16 + vc) * HD + vd0);
            r1.q = *(const u32x4*)(vsrc + (1 * 16 + vc) * HD + vd0);
            r2.q = *(const u32x4*)(vsrc + (2 * 16 + vc) * HD + vd0);
            r3.q = *(const u32x4*)(vsrc + (3 * 16 + vc) * HD + vd0);
            #pragma unroll
            for (int d = 0; d < 8; ++d) {
                unsigned long long pk =
                      (unsigned long long)r0.h[d]
                    | ((unsigned long long)r1.h[d] << 16)
                    | ((unsigned long long)r2.h[d] << 32)
                    | ((unsigned long long)r3.h[d] << 48);
                *(unsigned long long*)(Vts + (vd0 + d) * 72 + 4 * vc) = pk;
            }
        }
        // prefetch next chunk's K and V cachelines
        if (ch + 1 < SEQ / 64) {
            __builtin_prefetch(kbase + (size_t)(ch + 1) * 64 * HD + tid * 32, 0, 3);
            __builtin_prefetch(vbase + (size_t)(ch + 1) * 64 * HD + tid * 32, 0, 3);
        }
#ifdef HAVE_ASYNC_LDS
#if __has_builtin(__builtin_amdgcn_s_wait_asynccnt)
        __builtin_amdgcn_s_wait_asynccnt(0);
#else
        asm volatile("s_wait_asynccnt 0x0" ::: "memory");
#endif
#endif
        __syncthreads();

        // ---- S = Q K^T over 64 keys (four 16-key N tiles) ----------------
        v8f s[4];
        #pragma unroll
        for (int t = 0; t < 4; ++t) { v8f z = {}; s[t] = z; }
        #pragma unroll
        for (int dc = 0; dc < 2; ++dc) {
            int kr = dc * 32 + half * 8;
            #pragma unroll
            for (int t = 0; t < 4; ++t) {
                const unsigned short* p = Ks + (t * 16 + l15) * 64 + kr;
                UF16 tb;
                tb.q[0] = *(const u32x4*)p;
                tb.q[1] = *(const u32x4*)(p + 16);
                s[t] = wmma_bf16(qa[dc], tb.v, s[t]);
            }
        }

        // ---- online softmax (scale pre-folded into Q); packed b64 P store
        #pragma unroll
        for (int i = 0; i < 8; ++i) {
            float v0 = s[0][i], v1 = s[1][i], v2 = s[2][i], v3 = s[3][i];
            float rm = fmaxf(fmaxf(v0, v1), fmaxf(v2, v3));
            #pragma unroll
            for (int off = 8; off >= 1; off >>= 1) rm = fmaxf(rm, __shfl_xor(rm, off, 32));
            float nm   = fmaxf(m[i], rm);
            float corr = __expf(m[i] - nm);
            m[i] = nm;
            lsum[i] *= corr;
            o[0][i] *= corr; o[1][i] *= corr; o[2][i] *= corr; o[3][i] *= corr;
            int row = half * 8 + i;
            unsigned long long pk =
                  (unsigned long long)f2bf(__expf(v0 - nm))
                | ((unsigned long long)f2bf(__expf(v1 - nm)) << 16)
                | ((unsigned long long)f2bf(__expf(v2 - nm)) << 32)
                | ((unsigned long long)f2bf(__expf(v3 - nm)) << 48);
            *(unsigned long long*)(Ps + wv * 1152 + row * 72 + 4 * l15) = pk;
        }

        // ---- P re-layout C->A through per-wave LDS (wave-local, in-order DS)
        us16 pa[2];
        #pragma unroll
        for (int c = 0; c < 2; ++c) {
            const unsigned short* pp = Ps + wv * 1152 + l15 * 72 + c * 32 + half * 8;
            UF16 t;
            t.q[0] = *(const u32x4*)pp;
            t.q[1] = *(const u32x4*)(pp + 16);
            pa[c] = t.v;
        }

        // ---- row sums via WMMA against all-ones B ------------------------
        {
            v8f rs = {};
            rs = wmma_bf16(pa[0], ones, rs);
            rs = wmma_bf16(pa[1], ones, rs);
            #pragma unroll
            for (int i = 0; i < 8; ++i) lsum[i] += rs[i];
        }

        // ---- O += P * V : V B-fragments from transposed+permuted LDS tile
        #pragma unroll
        for (int kc = 0; kc < 2; ++kc) {
            #pragma unroll
            for (int ng = 0; ng < 4; ++ng) {
                const unsigned short* vp = Vts + (ng * 16 + l15) * 72 + kc * 32 + half * 8;
                UF16 t;
                t.q[0] = *(const u32x4*)vp;
                t.q[1] = *(const u32x4*)(vp + 16);
                o[ng] = wmma_bf16(pa[kc], t.v, o[ng]);
            }
        }
    }

    // ---- normalize and store fp32 O: [bh][seq][64] -----------------------
    float* obase = oh + ((size_t)bh * SEQ + qblk * 64 + wv * 16) * HD;
    #pragma unroll
    for (int i = 0; i < 8; ++i) {
        float inv = 1.0f / lsum[i];
        int row = half * 8 + i;
        #pragma unroll
        for (int ng = 0; ng < 4; ++ng)
            obase[row * HD + ng * 16 + l15] = o[ng][i] * inv;
    }
}

// ---------------------------------------------------------------------------
// Kernel 3: merge heads + output Tucker projection, fp32 out
// ---------------------------------------------------------------------------
__global__ void __launch_bounds__(256) out_proj_kernel(
    const float* __restrict__ oh,
    const float* __restrict__ w1o, const float* __restrict__ w2o, const float* __restrict__ w3o,
    float* __restrict__ out)
{
    __shared__ float xv[EV], t1[EV], t2[EV];
    __shared__ float ws1[64], ws2[64], ws3[144];

    const int vox = blockIdx.x;
    const int b   = vox >> 10;
    const int seq = vox & 1023;
    const int tid = threadIdx.x;

    // gather merged heads: e1 = x*2+h1, e2 = y*2+h2, e3 = z*3+h3
    for (int e = tid; e < EV; e += 256) {
        int e1 = e / 96, r = e - e1 * 96;
        int e2 = r / 12, e3 = r - e2 * 12;
        int hx = e1 >> 1, h1 = e1 & 1;
        int hy = e2 >> 1, h2 = e2 & 1;
        int hz = e3 / 3,  h3 = e3 - hz * 3;
        int head = (h1 * 2 + h2) * 3 + h3;
        int d = hx * 16 + hy * 4 + hz;
        xv[e] = oh[(((size_t)(b * NH + head) * SEQ) + seq) * HD + d];
    }
    if (tid < 64)  { ws1[tid] = w1o[tid]; ws2[tid] = w2o[tid]; }
    if (tid < 144) { ws3[tid] = w3o[tid]; }
    __syncthreads();

    for (int e = tid; e < EV; e += 256) {
        int xx = e / 96, rem = e - xx * 96;
        float acc = 0.f;
        #pragma unroll
        for (int h = 0; h < 8; ++h) acc += xv[h * 96 + rem] * ws1[h * 8 + xx];
        t1[e] = acc;
    }
    __syncthreads();
    for (int e = tid; e < EV; e += 256) {
        int xx = e / 96, r = e - xx * 96;
        int yy = r / 12, c = r - yy * 12;
        float acc = 0.f;
        #pragma unroll
        for (int w = 0; w < 8; ++w) acc += t1[xx * 96 + w * 12 + c] * ws2[w * 8 + yy];
        t2[e] = acc;
    }
    __syncthreads();
    for (int e = tid; e < EV; e += 256) {
        int e1 = e / 96, r = e - e1 * 96;
        int e2 = r / 12, e3 = r - e2 * 12;
        float acc = 0.f;
        #pragma unroll
        for (int c = 0; c < 12; ++c) acc += t2[e1 * 96 + e2 * 12 + c] * ws3[c * 12 + e3];
        out[(size_t)vox * EV + e] = acc;
    }
}

// ---------------------------------------------------------------------------
extern "C" void kernel_launch(void* const* d_in, const int* in_sizes, int n_in,
                              void* d_out, int out_size, void* d_ws, size_t ws_size,
                              hipStream_t stream) {
    const float* x   = (const float*)d_in[0];
    const float* w1q = (const float*)d_in[1];
    const float* w2q = (const float*)d_in[2];
    const float* w3q = (const float*)d_in[3];
    const float* w1k = (const float*)d_in[4];
    const float* w2k = (const float*)d_in[5];
    const float* w3k = (const float*)d_in[6];
    const float* w1v = (const float*)d_in[7];
    const float* w2v = (const float*)d_in[8];
    const float* w3v = (const float*)d_in[9];
    const float* w1o = (const float*)d_in[10];
    const float* w2o = (const float*)d_in[11];
    const float* w3o = (const float*)d_in[12];

    char* ws = (char*)d_ws;
    unsigned short* qh = (unsigned short*)ws;
    unsigned short* kh = qh + NQKV;
    unsigned short* vh = kh + NQKV;
    float* oh = (float*)(ws + (size_t)3 * NQKV * 2);

    proj_qkv_kernel<<<dim3(NB * SEQ), dim3(256), 0, stream>>>(
        x, w1q, w2q, w3q, w1k, w2k, w3k, w1v, w2v, w3v, qh, kh, vh);
    attn_kernel<<<dim3(NB * NH, SEQ / 64), dim3(128), 0, stream>>>(qh, kh, vh, oh);
    out_proj_kernel<<<dim3(NB * SEQ), dim3(256), 0, stream>>>(
        oh, w1o, w2o, w3o, (float*)d_out);
}
